// HashedMLP_42339787604520
// MI455X (gfx1250) — matrix operand order repaired and checked by reading, not compiled, not run on hardware
//
#include <hip/hip_runtime.h>
#include <hip/hip_bf16.h>

#define N_LEVEL   13
#define N_ENTRIES (1u << 19)
#define HASH_MASK (N_ENTRIES - 1u)
#define HDIM      64

typedef _Float16 v16h __attribute__((ext_vector_type(16)));
typedef float    v8f  __attribute__((ext_vector_type(8)));

union HFrag { v16h h; unsigned u[8]; };

__constant__ float c_res[N_LEVEL] = {
    16.f, 23.f, 32.f, 45.f, 64.f, 91.f, 128.f,
    181.f, 256.f, 362.f, 512.f, 724.f, 1024.f
};

__device__ __forceinline__ unsigned pack2h(float a, float b) {
    union { _Float16 h[2]; unsigned u; } x;
    x.h[0] = (_Float16)a; x.h[1] = (_Float16)b;
    return x.u;
}

// branch-free tanh-approx GELU (jax.nn.gelu default)
__device__ __forceinline__ float gelu_tanh(float x) {
    float u = 0.7978845608028654f * x + 0.03567740814183287f * (x * x * x);
#if __has_builtin(__builtin_amdgcn_tanhf)
    return 0.5f * x * (1.0f + __builtin_amdgcn_tanhf(u));
#elif __has_builtin(__builtin_amdgcn_tanh_f32)
    return 0.5f * x * (1.0f + __builtin_amdgcn_tanh_f32(u));
#else
    // 1 + tanh(u) = 2 * (1 - 1/(1+e^{2u}));  e^{2u} = exp2(u * 2*log2(e))
    float e = __builtin_amdgcn_exp2f(u * 2.885390081777927f);
    return x * (1.0f - __builtin_amdgcn_rcpf(1.0f + e));
#endif
}

// one wave (32 lanes) processes one 16-point tile per loop iteration
__global__ __launch_bounds__(256) void HashedMLP_42339787604520_kernel(
    const float* __restrict__ data,   // [N,3]
    const float* __restrict__ table,  // [13, 2^19, 2]
    const float* __restrict__ W0,     // [26,64]
    const float* __restrict__ b0,     // [64]
    const float* __restrict__ W1,     // [64,64]
    const float* __restrict__ b1,     // [64]
    const float* __restrict__ W2,     // [64,1]
    const float* __restrict__ b2,     // [1]
    float* __restrict__ out,          // [N,1]
    int nPoints)
{
    __shared__ unsigned  sF [8][16][16];   // per-wave 16x32 f16 features (packed u32)
    __shared__ _Float16  sH1[8][16][64];   // per-wave 16x64 f16 hidden-0
    __shared__ float     sP [8][48];       // per-wave 16 points x 3 coords

    const int tid  = threadIdx.x;
    const int lane = tid & 31;
    const int w    = tid >> 5;             // wave id in block (0..7)
    const int n16  = lane & 15;            // N-column / M-row within tile
    const int hi   = lane >> 4;            // half-wave select
    const int hi4  = hi * 4;               // unused (A k-offset in uints is hi8)
    const int hi8  = hi * 8;

    // ---- preload f16 weight fragments into VGPRs (held across tile loop) ----
    // B k-order used here (contiguous, same as A; B is 32x16): lane n (0..15) col N=n holds
    // K=0..15 packed 2/VGPR; lanes 16..31 hold K=16..31.
    HFrag bw0[4];                          // W0: K padded 26->32, 4 N-tiles
#pragma unroll
    for (int t = 0; t < 4; ++t) {
        int n = 16 * t + n16;
#pragma unroll
        for (int v = 0; v < 8; ++v) {
            int k = 2 * v + hi * 16;
            float f0 = (k     < 26) ? W0[k * HDIM + n]       : 0.0f;
            float f1 = (k + 1 < 26) ? W0[(k + 1) * HDIM + n] : 0.0f;
            bw0[t].u[v] = pack2h(f0, f1);
        }
    }
    HFrag bw1[2][4];                       // W1: 2 K-chunks x 4 N-tiles
#pragma unroll
    for (int kc = 0; kc < 2; ++kc) {
#pragma unroll
        for (int t = 0; t < 4; ++t) {
            int n = 16 * t + n16;
#pragma unroll
            for (int v = 0; v < 8; ++v) {
                int k = 32 * kc + 2 * v + hi * 16;
                bw1[kc][t].u[v] = pack2h(W1[k * HDIM + n], W1[(k + 1) * HDIM + n]);
            }
        }
    }
    float b0f[4], b1f[4], w2c[4];
#pragma unroll
    for (int t = 0; t < 4; ++t) {
        b0f[t] = b0[16 * t + n16];
        b1f[t] = b1[16 * t + n16];
        w2c[t] = W2[16 * t + n16];
    }
    const float b2v = b2[0];

    // K-padding (cols 26..31) of the feature tile: written once, never dirtied
    if (lane < 16) {
        sF[w][lane][13] = 0u; sF[w][lane][14] = 0u; sF[w][lane][15] = 0u;
    }

    const int nTiles   = nPoints >> 4;
    const int gwave    = blockIdx.x * 8 + w;
    const int totWaves = gridDim.x * 8;

    for (int tile = gwave; tile < nTiles; tile += totWaves) {
        const int tbase = tile * 16;

        // ---- stage 16 points' coords (2 coalesced loads) --------------------
#pragma unroll
        for (int i = lane; i < 48; i += 32)
            sP[w][i] = data[tbase * 3 + i];

        // ---- hash-grid features: 16 pts x 13 levels = 208 (pt,lvl) pairs ----
        for (int it = 0; it < 7; ++it) {
            int p = it * 32 + lane;
            if (p < 208) {
                int m = p / 13;
                int l = p - m * 13;
                float r  = c_res[l];
                float px = sP[w][m * 3 + 0] * r;
                float py = sP[w][m * 3 + 1] * r;
                float pz = sP[w][m * 3 + 2] * r;
                float gx = floorf(px), gy = floorf(py), gz = floorf(pz);
                float fx = px - gx, fy = py - gy, fz = pz - gz;
                unsigned ix = (unsigned)(int)gx;
                unsigned iy = (unsigned)(int)gy;
                unsigned iz = (unsigned)(int)gz;
                const float* tb = table + ((size_t)l << 20);  // l * 2^19 * 2
                float f0 = 0.f, f1 = 0.f;
#pragma unroll
                for (int c = 0; c < 8; ++c) {
                    unsigned cx = ix + (c & 1);
                    unsigned cy = iy + ((c >> 1) & 1);
                    unsigned cz = iz + ((c >> 2) & 1);
                    unsigned h  = (cx ^ (cy * 2654435761u) ^ (cz * 805459861u)) & HASH_MASK;
                    float wx = (c & 1)        ? fx : 1.f - fx;
                    float wy = ((c >> 1) & 1) ? fy : 1.f - fy;
                    float wz = ((c >> 2) & 1) ? fz : 1.f - fz;
                    float wt = wx * wy * wz;
                    float2 fv = *(const float2*)(tb + 2 * (size_t)h);
                    f0 += wt * fv.x;
                    f1 += wt * fv.y;
                }
                sF[w][m][l] = pack2h(f0, f1);
            }
        }

        // ---- layer 0: [16x32] @ [32x64] via 4 WMMAs -------------------------
        // A layout 16x32 f16: lane m=lane&15; uint idx = v + hi*8
        HFrag a0;
#pragma unroll
        for (int v = 0; v < 8; ++v)
            a0.u[v] = sF[w][n16][v + hi8];

        v8f acc0[4];
#pragma unroll
        for (int t = 0; t < 4; ++t) {
            v8f c;
#pragma unroll
            for (int r = 0; r < 8; ++r) c[r] = b0f[t];
            acc0[t] = __builtin_amdgcn_wmma_f32_16x16x32_f16(
                false, a0.h, false, bw0[t].h, (short)0, c, false, false);
        }

        // GELU + restage 16x64 f16 into LDS (D layout: lane holds rows r+8*hi, col 16t+n16)
#pragma unroll
        for (int t = 0; t < 4; ++t) {
#pragma unroll
            for (int r = 0; r < 8; ++r)
                sH1[w][r + hi8][16 * t + n16] = (_Float16)gelu_tanh(acc0[t][r]);
        }

        // ---- layer 1: [16x64] @ [64x64] via 2x4 WMMAs -----------------------
        v8f acc1[4];
#pragma unroll
        for (int t = 0; t < 4; ++t) {
            v8f c;
#pragma unroll
            for (int r = 0; r < 8; ++r) c[r] = b1f[t];
            acc1[t] = c;
        }
        const unsigned* h1u = (const unsigned*)&sH1[w][0][0];
#pragma unroll
        for (int kc = 0; kc < 2; ++kc) {
            HFrag a1;
#pragma unroll
            for (int v = 0; v < 8; ++v)
                a1.u[v] = h1u[n16 * 32 + 16 * kc + v + hi8];
#pragma unroll
            for (int t = 0; t < 4; ++t)
                acc1[t] = __builtin_amdgcn_wmma_f32_16x16x32_f16(
                    false, a1.h, false, bw1[kc][t].h, (short)0, acc1[t], false, false);
        }

        // ---- layer 2: 64 -> 1. Fold W2 per-lane, butterfly-reduce over the
        //      16-lane half (columns), lanes n16==0 hold the 16 row outputs. --
        float partial[8];
#pragma unroll
        for (int r = 0; r < 8; ++r) {
            float s = 0.f;
#pragma unroll
            for (int t = 0; t < 4; ++t)
                s += gelu_tanh(acc1[t][r]) * w2c[t];
            partial[r] = s;
        }
#pragma unroll
        for (int mask = 1; mask <= 8; mask <<= 1) {
#pragma unroll
            for (int r = 0; r < 8; ++r)
                partial[r] += __shfl_xor(partial[r], mask, 32);
        }
        if (n16 == 0) {
#pragma unroll
            for (int r = 0; r < 8; ++r)
                out[tbase + r + hi8] = partial[r] + b2v;
        }
    }
}

extern "C" void kernel_launch(void* const* d_in, const int* in_sizes, int n_in,
                              void* d_out, int out_size, void* d_ws, size_t ws_size,
                              hipStream_t stream) {
    const float* data  = (const float*)d_in[0];
    const float* table = (const float*)d_in[1];
    const float* W0    = (const float*)d_in[2];
    const float* b0    = (const float*)d_in[3];
    const float* W1    = (const float*)d_in[4];
    const float* b1    = (const float*)d_in[5];
    const float* W2    = (const float*)d_in[6];
    const float* b2    = (const float*)d_in[7];
    float* out = (float*)d_out;

    int nPoints = in_sizes[0] / 3;   // 524288
    // 512 blocks x 8 waves = 4096 waves; 8 tiles of 16 points per wave
    dim3 grid(512), block(256);
    HashedMLP_42339787604520_kernel<<<grid, block, 0, stream>>>(
        data, table, W0, b0, W1, b1, W2, b2, out, nPoints);
}
